// Codebook_5961414607133
// MI455X (gfx1250) — matrix operand order repaired and checked by reading, not compiled, not run on hardware
//
#include <hip/hip_runtime.h>

typedef __attribute__((ext_vector_type(16))) _Float16 v16h;
typedef __attribute__((ext_vector_type(8)))  _Float16 v8h;
typedef __attribute__((ext_vector_type(4)))  _Float16 v4h;
typedef __attribute__((ext_vector_type(8)))  float    v8f;

#define B_    32
#define C_    256
#define S_    1024              // H*W
#define K_    1024
#define L_    64
#define OUT0  (B_ * C_ * S_)    // 8388608 floats, then nearest [B_*S_, L_]

// workspace layout (byte offsets, all 16B aligned)
#define WS_CB16    0                       // K_*L_ halves   = 131072 B
#define WS_WIN16   (WS_CB16   + K_*L_*2)   // L_*C_ halves   =  32768 B
#define WS_WOUT16  (WS_WIN16  + L_*C_*2)   // C_*L_ halves   =  32768 B
#define WS_HNORM   (WS_WOUT16 + C_*L_*2)   // K_ floats      =   4096 B

// ---------------------------------------------------------------------------
// Fragment loaders.
// A (16x32 f16, MxK): lane half h: v16h elems 0..7  = K [kc+8h .. kc+8h+7],
//                                       elems 8..15 = K [kc+16+8h .. +7]
// B (32x16 f16, KxN): lane half h: v16h elems 0..7  = K [kc+8h .. kc+8h+7],
//                                       elems 8..15 = K [kc+16+8h .. +7],
//                     column n = lane%16 (column-contiguous source rows)
// ---------------------------------------------------------------------------
__device__ __forceinline__ v16h load_a_frag(const _Float16* base, int ldk,
                                            int m, int kc, int half) {
  union { v16h v; v8h h[2]; } u;
  const _Float16* p = base + m * ldk + kc + half * 8;
  u.h[0] = *(const v8h*)(p);
  u.h[1] = *(const v8h*)(p + 16);
  return u.v;
}

__device__ __forceinline__ v16h load_b_frag(const _Float16* colbase, int kc, int half) {
  union { v16h v; v8h h[2]; } u;
  const _Float16* p = colbase + kc + half * 8;
  u.h[0] = *(const v8h*)(p);
  u.h[1] = *(const v8h*)(p + 16);
  return u.v;
}

// ---------------------------------------------------------------------------
// Prep kernels: f16 copies + 0.5*||c||^2
// ---------------------------------------------------------------------------
__global__ void prep_cb(const float* __restrict__ cb, _Float16* __restrict__ cb16,
                        float* __restrict__ hnorm) {
  const int code = blockIdx.x;          // one wave32 per code
  const int lane = threadIdx.x;
  const float* row = cb + code * L_;
  float a = row[lane];
  float b = row[lane + 32];
  cb16[code * L_ + lane]      = (_Float16)a;
  cb16[code * L_ + lane + 32] = (_Float16)b;
  float s = a * a + b * b;
  for (int m = 16; m; m >>= 1) s += __shfl_xor(s, m, 32);
  if (lane == 0) hnorm[code] = 0.5f * s;
}

__global__ void prep_w(const float* __restrict__ Win, const float* __restrict__ Wout,
                       _Float16* __restrict__ Win16, _Float16* __restrict__ Wout16) {
  const int i = blockIdx.x * blockDim.x + threadIdx.x;   // 16384 each
  Win16[i]  = (_Float16)Win[i];    // [L_][C_] flat copy
  Wout16[i] = (_Float16)Wout[i];   // [C_][L_] flat copy
}

// ---------------------------------------------------------------------------
// Fused VQ kernel: one block (8 waves) per 16 spatial rows.
// ---------------------------------------------------------------------------
__global__ void __launch_bounds__(256) vq_fused(
    const float*    __restrict__ x,       // [B_, C_, S_]
    const float*    __restrict__ cb,      // [K_, L_] fp32
    const float*    __restrict__ b_in,    // [L_]
    const float*    __restrict__ b_out,   // [C_]
    const _Float16* __restrict__ cb16,    // [K_][L_]
    const _Float16* __restrict__ Win16,   // [L_][C_]
    const _Float16* __restrict__ Wout16,  // [C_][L_]
    const float*    __restrict__ hnorm,   // [K_]
    float*          __restrict__ out,     // [B_, C_, S_]
    float*          __restrict__ nearest) // [B_*S_, L_]
{
  __shared__ __align__(16) _Float16 xA[16][C_];   // x tile, f16
  __shared__ __align__(16) _Float16 zA[16][L_];   // proj_in result, f16
  __shared__ __align__(16) _Float16 nA[16][L_];   // gathered codes, f16
  __shared__ float redV[8][16];
  __shared__ int   redI[8][16];
  __shared__ int   bestRow[16];

  const int tid  = threadIdx.x;
  const int wave = tid >> 5;
  const int lane = tid & 31;
  const int half = lane >> 4;
  const int nl   = lane & 15;

  const int tile = blockIdx.x;           // 0..2047
  const int sg0  = tile * 16;            // global flat row base (b*S_ + s)
  const int b    = sg0 >> 10;            // S_ == 1024
  const int s0   = sg0 & (S_ - 1);

  // ---- phase 0: stage x tile -> LDS (f16), coalesced along s ----
  {
    const int m  = tid & 15;
    const int k0 = (tid >> 4) * 16;
    const float* xp = x + (size_t)b * C_ * S_ + s0 + m;
    for (int kk = 0; kk < 16; ++kk)
      xA[m][k0 + kk] = (_Float16)xp[(size_t)(k0 + kk) * S_];
  }
  __syncthreads();

  // ---- phase 1: z = x_tile * W_in^T + b_in  (waves 0..3, one 16x16 N-tile each)
  if (wave < 4) {
    const int lcol = wave * 16 + nl;
    const _Float16* bcol = Win16 + lcol * C_;   // row of W_in == column of B
    v8f acc = {};
    for (int kc = 0; kc < C_; kc += 32) {
      v16h a  = load_a_frag(&xA[0][0], C_, nl, kc, half);
      v16h bb = load_b_frag(bcol, kc, half);
      acc = __builtin_amdgcn_wmma_f32_16x16x32_f16(false, a, false, bb,
                                                   (short)0, acc, false, false);
    }
    const float bias = b_in[lcol];
    for (int v = 0; v < 8; ++v)
      zA[v + half * 8][lcol] = (_Float16)(acc[v] + bias);
  }
  __syncthreads();

  // ---- phase 2: scores = z * cb^T; argmin_k (0.5*||c||^2 - z.c) ----
  {
    float bv[8];
    int   bi[8];
    for (int v = 0; v < 8; ++v) { bv[v] = 3.4e38f; bi[v] = 0; }
    for (int j = 0; j < 8; ++j) {
      const int code = (wave * 8 + j) * 16 + nl;
      const _Float16* bcol = cb16 + code * L_;
      v8f acc = {};
      for (int kc = 0; kc < L_; kc += 32) {
        v16h a  = load_a_frag(&zA[0][0], L_, nl, kc, half);
        v16h bb = load_b_frag(bcol, kc, half);
        acc = __builtin_amdgcn_wmma_f32_16x16x32_f16(false, a, false, bb,
                                                     (short)0, acc, false, false);
      }
      const float hn = hnorm[code];
      for (int v = 0; v < 8; ++v) {
        const float val = hn - acc[v];
        if (val < bv[v] || (val == bv[v] && code < bi[v])) { bv[v] = val; bi[v] = code; }
      }
    }
    // reduce across the 16 N-lanes of each half (rows stay within half)
    for (int m = 1; m < 16; m <<= 1) {
      for (int v = 0; v < 8; ++v) {
        const float ov = __shfl_xor(bv[v], m, 32);
        const int   oi = __shfl_xor(bi[v], m, 32);
        if (ov < bv[v] || (ov == bv[v] && oi < bi[v])) { bv[v] = ov; bi[v] = oi; }
      }
    }
    if (nl == 0) {
      for (int v = 0; v < 8; ++v) {
        redV[wave][v + half * 8] = bv[v];
        redI[wave][v + half * 8] = bi[v];
      }
    }
  }
  __syncthreads();
  if (tid < 16) {
    float best = redV[0][tid];
    int   bidx = redI[0][tid];
    for (int w = 1; w < 8; ++w) {
      const float ov = redV[w][tid];
      const int   oi = redI[w][tid];
      if (ov < best || (ov == best && oi < bidx)) { best = ov; bidx = oi; }
    }
    bestRow[tid] = bidx;
  }
  __syncthreads();

  // ---- phase 3: gather nearest (fp32 exact to output; f16 copy for GEMM3) ----
  {
    const int m = tid >> 4;
    const int l = (tid & 15) * 4;
    const int code = bestRow[m];
    const float4 q = *(const float4*)(cb + code * L_ + l);
    *(float4*)(nearest + (size_t)(sg0 + m) * L_ + l) = q;
    v4h h4;
    h4[0] = (_Float16)q.x; h4[1] = (_Float16)q.y;
    h4[2] = (_Float16)q.z; h4[3] = (_Float16)q.w;
    *(v4h*)(&nA[m][l]) = h4;
  }
  __syncthreads();

  // ---- phase 4: out = nearest * W_out^T + b_out, stored transposed [B,C,S] ----
  for (int t = 0; t < 2; ++t) {
    const int c = (wave * 2 + t) * 16 + nl;
    const _Float16* bcol = Wout16 + c * L_;     // row of W_out == column of B
    v8f acc = {};
    for (int kc = 0; kc < L_; kc += 32) {
      v16h a  = load_a_frag(&nA[0][0], L_, nl, kc, half);
      v16h bb = load_b_frag(bcol, kc, half);
      acc = __builtin_amdgcn_wmma_f32_16x16x32_f16(false, a, false, bb,
                                                   (short)0, acc, false, false);
    }
    const float bias = b_out[c];
    // D layout: lane -> column c, VGPR v -> row m = v + 8*half (consecutive s)
    float* op = out + (size_t)b * C_ * S_ + (size_t)c * S_ + s0 + half * 8;
    float4 lo = make_float4(acc[0] + bias, acc[1] + bias, acc[2] + bias, acc[3] + bias);
    float4 hi = make_float4(acc[4] + bias, acc[5] + bias, acc[6] + bias, acc[7] + bias);
    *(float4*)(op)     = lo;
    *(float4*)(op + 4) = hi;
  }
}

// ---------------------------------------------------------------------------
extern "C" void kernel_launch(void* const* d_in, const int* in_sizes, int n_in,
                              void* d_out, int out_size, void* d_ws, size_t ws_size,
                              hipStream_t stream) {
  const float* x    = (const float*)d_in[0];
  const float* cb   = (const float*)d_in[1];
  const float* Win  = (const float*)d_in[2];
  const float* bin  = (const float*)d_in[3];
  const float* Wout = (const float*)d_in[4];
  const float* bout = (const float*)d_in[5];

  char* ws = (char*)d_ws;
  _Float16* cb16   = (_Float16*)(ws + WS_CB16);
  _Float16* Win16  = (_Float16*)(ws + WS_WIN16);
  _Float16* Wout16 = (_Float16*)(ws + WS_WOUT16);
  float*    hnorm  = (float*)(ws + WS_HNORM);

  float* out     = (float*)d_out;
  float* nearest = out + OUT0;

  prep_cb<<<K_, 32, 0, stream>>>(cb, cb16, hnorm);
  prep_w<<<(L_ * C_) / 256, 256, 0, stream>>>(Win, Wout, Win16, Wout16);
  vq_fused<<<(B_ * S_) / 16, 256, 0, stream>>>(x, cb, bin, bout,
                                               cb16, Win16, Wout16, hnorm,
                                               out, nearest);
}
